// IPEXTransformerAtten_32255204393424
// MI455X (gfx1250) — matrix operand order repaired and checked by reading, not compiled, not run on hardware
//
#include <hip/hip_runtime.h>

// ---------------------------------------------------------------------------
// CDNA5 (gfx1250) causal self-attention, bf16 WMMA pipeline, f32 accumulate.
//   f32 inputs --cvt--> bf16 (weights transposed to [n][k])
//   Q/K = hidden @ W (row-major bf16 out), V = hidden @ Wv stored transposed
//   flash attention per head (QK^T, online softmax, PV), all WMMA
//   out = attn @ Wo^T-layout -> f32 d_out
// Global->LDS staging uses async loads (gfx1250 GLOBAL_LOAD_ASYNC_TO_LDS_B128)
// with double-buffered LDS and one barrier per K-step.
// ---------------------------------------------------------------------------

#define USE_ASYNC_LDS 1

typedef __attribute__((ext_vector_type(16))) __bf16 bf16x16;
typedef __attribute__((ext_vector_type(8)))  float  f32x8;

constexpr int kS = 2048;   // sequence
constexpr int kE = 2048;   // embed
constexpr int kH = 16;     // heads
constexpr int kD = 128;    // head dim
constexpr float kMaskNeg = -66504.0f;
constexpr float kScale   = 0.08838834764831845f;  // 1/sqrt(128)

__device__ __forceinline__ unsigned short f32_to_bf16(float f) {
  unsigned u = __float_as_uint(f);
  u += 0x7fffu + ((u >> 16) & 1u);   // round-to-nearest-even
  return (unsigned short)(u >> 16);
}

// 16-byte global->LDS copy. Async path: LDS byte offset is the low 32 bits of
// the flat shared-aperture address (ISA 10.2).
__device__ __forceinline__ void copy16_g2l(const unsigned short* __restrict__ g,
                                           unsigned short* __restrict__ l) {
#if USE_ASYNC_LDS
  unsigned lds_off = (unsigned)(size_t)l;
  asm volatile("global_load_async_to_lds_b128 %0, %1, off"
               :: "v"(lds_off), "v"(g) : "memory");
#else
  *reinterpret_cast<uint4*>(l) = *reinterpret_cast<const uint4*>(g);
#endif
}

__device__ __forceinline__ void async_wait0() {
#if USE_ASYNC_LDS
  asm volatile("s_wait_asynccnt 0x0" ::: "memory");
#endif
}

// ---------------- f32 -> bf16 conversion (row-major) -----------------------
__global__ __launch_bounds__(256) void cvt_f32_bf16(const float* __restrict__ src,
                                                    unsigned short* __restrict__ dst) {
  size_t i = ((size_t)blockIdx.x * 256 + threadIdx.x) * 4;
  float4 v = *reinterpret_cast<const float4*>(src + i);
  uint2 pkt;
  pkt.x = (unsigned)f32_to_bf16(v.x) | ((unsigned)f32_to_bf16(v.y) << 16);
  pkt.y = (unsigned)f32_to_bf16(v.z) | ((unsigned)f32_to_bf16(v.w) << 16);
  *reinterpret_cast<uint2*>(dst + i) = pkt;
}

// ---------------- f32 [k][n] -> bf16 transposed [n][k] ---------------------
__global__ __launch_bounds__(256) void cvt_transpose_f32_bf16(
    const float* __restrict__ src, unsigned short* __restrict__ dst) {
  __shared__ float tile[32][33];
  const int r0 = blockIdx.y * 32, c0 = blockIdx.x * 32;
  const int tc = threadIdx.x & 31, tr = threadIdx.x >> 5;
#pragma unroll
  for (int j = 0; j < 4; ++j) {
    int row = tr + j * 8;
    tile[row][tc] = src[(size_t)(r0 + row) * kE + c0 + tc];
  }
  __syncthreads();
#pragma unroll
  for (int j = 0; j < 2; ++j) {
    int idx  = threadIdx.x + j * 256;
    int orow = idx >> 4, pr = idx & 15;       // output row n=c0+orow, cols m
    unsigned lo = f32_to_bf16(tile[pr * 2][orow]);
    unsigned hi = f32_to_bf16(tile[pr * 2 + 1][orow]);
    reinterpret_cast<unsigned*>(dst)[(((size_t)(c0 + orow) * kE) + r0) / 2 + pr] =
        lo | (hi << 16);
  }
}

// ---------------- bf16 GEMM: C = A[m][k] @ Bt[n][k]^T ----------------------
// 128x128 block, 8 waves as 4(m) x 2(n); wave tile 32x64 = 2x4 WMMA tiles.
// Double-buffered LDS, async global->LDS, 12 ds_load_b128 : 8 WMMA per step.
// OUT_MODE: 0 = bf16 row-major, 1 = f32 row-major, 2 = bf16 transposed [n][m]
template <int OUT_MODE>
__global__ __launch_bounds__(256) void gemm_bf16_wmma(
    const unsigned short* __restrict__ A, const unsigned short* __restrict__ Bt,
    void* __restrict__ C) {
  __shared__ unsigned short Abuf[2][128 * 32];   // [m][k]
  __shared__ unsigned short Bbuf[2][128 * 32];   // [n][k]

  const int tid  = threadIdx.x;
  const int wave = tid >> 5;
  const int lane = tid & 31;
  const int half = lane >> 4;
  const int lr   = lane & 15;
  const int bm   = blockIdx.x * 128;
  const int bn   = blockIdx.y * 128;
  const int wm   = (wave & 3) * 32;
  const int wn   = (wave >> 2) * 64;

  f32x8 acc[2][4];
  for (int mi = 0; mi < 2; ++mi)
    for (int ni = 0; ni < 4; ++ni)
      for (int i = 0; i < 8; ++i) acc[mi][ni][i] = 0.0f;

  // tile = 128 rows x 32 bf16 = 512 x 16B; 2 per thread per matrix
  auto issue = [&](int k0, int buf) {
#pragma unroll
    for (int j = 0; j < 2; ++j) {
      int idx = tid + j * 256;
      int row = idx >> 2, c4 = idx & 3;
      copy16_g2l(A  + (size_t)(bm + row) * kE + k0 + c4 * 8, &Abuf[buf][idx * 8]);
      copy16_g2l(Bt + (size_t)(bn + row) * kE + k0 + c4 * 8, &Bbuf[buf][idx * 8]);
    }
  };

  const int NT = kE / 32;   // 64 K-steps
  issue(0, 0);
  for (int i = 0; i < NT; ++i) {
    async_wait0();
    __syncthreads();
    if (i + 1 < NT) issue((i + 1) * 32, (i + 1) & 1);

    const unsigned* Au = reinterpret_cast<const unsigned*>(Abuf[i & 1]);
    const unsigned* Bu = reinterpret_cast<const unsigned*>(Bbuf[i & 1]);
    union { bf16x16 v; unsigned u[8]; } af[2], bfr[4];
#pragma unroll
    for (int mi = 0; mi < 2; ++mi) {
      int m = wm + mi * 16 + lr;
#pragma unroll
      for (int p = 0; p < 8; ++p) {           // A: K = lane-half*16 + e
        int k = 2 * p + 16 * half;
        af[mi].u[p] = Au[(m * 32 + k) >> 1];
      }
    }
#pragma unroll
    for (int ni = 0; ni < 4; ++ni) {
      int n = wn + ni * 16 + lr;
#pragma unroll
      for (int p = 0; p < 8; ++p) {           // B: K = lane-half*16 + e
        int k = 2 * p + 16 * half;
        bfr[ni].u[p] = Bu[(n * 32 + k) >> 1];
      }
    }
#pragma unroll
    for (int mi = 0; mi < 2; ++mi)
#pragma unroll
      for (int ni = 0; ni < 4; ++ni)
        acc[mi][ni] = __builtin_amdgcn_wmma_f32_16x16x32_bf16(
            false, af[mi].v, false, bfr[ni].v, (short)0, acc[mi][ni], false, false);
  }

#pragma unroll
  for (int mi = 0; mi < 2; ++mi) {
#pragma unroll
    for (int ni = 0; ni < 4; ++ni) {
      int n = bn + wn + ni * 16 + lr;
      int mb = bm + wm + mi * 16 + 8 * half;
      if (OUT_MODE == 2) {                     // packed transposed store: 1 b128
        uint4 pk;
        pk.x = (unsigned)f32_to_bf16(acc[mi][ni][0]) | ((unsigned)f32_to_bf16(acc[mi][ni][1]) << 16);
        pk.y = (unsigned)f32_to_bf16(acc[mi][ni][2]) | ((unsigned)f32_to_bf16(acc[mi][ni][3]) << 16);
        pk.z = (unsigned)f32_to_bf16(acc[mi][ni][4]) | ((unsigned)f32_to_bf16(acc[mi][ni][5]) << 16);
        pk.w = (unsigned)f32_to_bf16(acc[mi][ni][6]) | ((unsigned)f32_to_bf16(acc[mi][ni][7]) << 16);
        reinterpret_cast<uint4*>(C)[((size_t)n * kS + mb) >> 3] = pk;
      } else {
#pragma unroll
        for (int r = 0; r < 8; ++r) {
          if (OUT_MODE == 1)
            reinterpret_cast<float*>(C)[(size_t)(mb + r) * kE + n] = acc[mi][ni][r];
          else
            reinterpret_cast<unsigned short*>(C)[(size_t)(mb + r) * kE + n] =
                f32_to_bf16(acc[mi][ni][r]);
        }
      }
    }
  }
}

// ---------------- flash attention (per head, causal) -----------------------
// grid: (S/64, H), 128 threads = 4 waves, each wave 16 query rows.
// K staged [key][d], V (pre-transposed in global) staged [d][key]; all
// fragment reads contiguous. Double-buffered, async staging.
__global__ __launch_bounds__(128)
void flash_attn_bf16(const unsigned short* __restrict__ Qb,
                     const unsigned short* __restrict__ Kb,
                     const unsigned short* __restrict__ Vt,   // [E][S]
                     unsigned short* __restrict__ Ob) {
  __shared__ unsigned short Kt[2][32 * 128];    // [key][d]
  __shared__ unsigned short Vl[2][128 * 32];    // [d][key]
  __shared__ unsigned short Pt[4][16 * 32];     // per-wave P bounce [q][key]

  const int tid  = threadIdx.x;
  const int wave = tid >> 5;
  const int lane = tid & 31;
  const int half = lane >> 4;
  const int lr   = lane & 15;
  const int h    = blockIdx.y;
  const int q0   = blockIdx.x * 64;
  const int qw   = q0 + wave * 16;

  // Resident Q fragments: D=128 as 4 K-chunks of 32 (A-matrix layout)
  union { bf16x16 v; unsigned u[8]; } qf[4];
  {
    const unsigned* Qu =
        reinterpret_cast<const unsigned*>(Qb + (size_t)(qw + lr) * kE + h * kD);
#pragma unroll
    for (int c = 0; c < 4; ++c)
#pragma unroll
      for (int p = 0; p < 8; ++p) {
        int k = c * 32 + 2 * p + 16 * half;
        qf[c].u[p] = Qu[k >> 1];
      }
  }

  f32x8 o[8];
  for (int t = 0; t < 8; ++t)
    for (int i = 0; i < 8; ++i) o[t][i] = 0.0f;
  float mrow[8], lrow[8];
  for (int r = 0; r < 8; ++r) { mrow[r] = -3.0e38f; lrow[r] = 0.0f; }

  const unsigned* Pu = reinterpret_cast<const unsigned*>(Pt[wave]);

  auto issueKV = [&](int kb, int buf) {
    int kbase = kb * 32;
#pragma unroll
    for (int j = 0; j < 4; ++j) {
      int idx = tid + j * 128;
      int kr = idx >> 4, kc = idx & 15;        // K: [key][d], 16 x 16B per row
      copy16_g2l(Kb + (size_t)(kbase + kr) * kE + h * kD + kc * 8, &Kt[buf][idx * 8]);
      int vr = idx >> 2, vc = idx & 3;         // V^T: [d][key], 4 x 16B per row
      copy16_g2l(Vt + (size_t)(h * kD + vr) * kS + kbase + vc * 8, &Vl[buf][idx * 8]);
    }
  };

  const int nkb = q0 / 32 + 2;
  issueKV(0, 0);
  for (int kb = 0; kb < nkb; ++kb) {
    const int kbase = kb * 32;
    async_wait0();
    __syncthreads();
    if (kb + 1 < nkb) issueKV(kb + 1, (kb + 1) & 1);

    const unsigned* Ku = reinterpret_cast<const unsigned*>(Kt[kb & 1]);
    const unsigned* Vu = reinterpret_cast<const unsigned*>(Vl[kb & 1]);

    // scores: two 16x16 tiles over 32 keys; reduce D in 4 chunks
    f32x8 sc[2];
    for (int kt = 0; kt < 2; ++kt)
      for (int i = 0; i < 8; ++i) sc[kt][i] = 0.0f;
#pragma unroll
    for (int c = 0; c < 4; ++c) {
#pragma unroll
      for (int kt = 0; kt < 2; ++kt) {
        union { bf16x16 v; unsigned u[8]; } kf;  // B = K^T: row of K contiguous
        int key = kt * 16 + lr;
#pragma unroll
        for (int p = 0; p < 8; ++p) {
          int d = c * 32 + 2 * p + 16 * half;
          kf.u[p] = Ku[(key * kD + d) >> 1];
        }
        sc[kt] = __builtin_amdgcn_wmma_f32_16x16x32_bf16(
            false, qf[c].v, false, kf.v, (short)0, sc[kt], false, false);
      }
    }

    // causal mask (add MASK_NEG then scale, matching reference) + online softmax
#pragma unroll
    for (int r = 0; r < 8; ++r) {
      const int qg = qw + r + 8 * half;
      float rm = -3.0e38f;
#pragma unroll
      for (int kt = 0; kt < 2; ++kt) {
        int kg = kbase + kt * 16 + lr;
        float s = sc[kt][r];
        if (kg > qg) s += kMaskNeg;
        s *= kScale;
        sc[kt][r] = s;
        rm = fmaxf(rm, s);
      }
#pragma unroll
      for (int off = 1; off < 16; off <<= 1)
        rm = fmaxf(rm, __shfl_xor(rm, off, 16));
      float mn    = fmaxf(mrow[r], rm);
      float alpha = __expf(mrow[r] - mn);
      float rl = 0.0f;
#pragma unroll
      for (int kt = 0; kt < 2; ++kt) {
        float pexp = __expf(sc[kt][r] - mn);
        sc[kt][r] = pexp;
        rl += pexp;
      }
#pragma unroll
      for (int off = 1; off < 16; off <<= 1)
        rl += __shfl_xor(rl, off, 16);
      lrow[r] = lrow[r] * alpha + rl;
      mrow[r] = mn;
#pragma unroll
      for (int t = 0; t < 8; ++t) o[t][r] *= alpha;
    }

    // P: C/D layout -> LDS bounce -> A-matrix layout fragment (16x32 bf16)
#pragma unroll
    for (int kt = 0; kt < 2; ++kt)
#pragma unroll
      for (int r = 0; r < 8; ++r)
        Pt[wave][(r + 8 * half) * 32 + kt * 16 + lr] = f32_to_bf16(sc[kt][r]);

    union { bf16x16 v; unsigned u[8]; } pf;
#pragma unroll
    for (int p = 0; p < 8; ++p) {
      int k = 2 * p + 16 * half;
      pf.u[p] = Pu[(lr * 32 + k) >> 1];
    }

    // O += P @ V : V fragment from [d][key] tile, contiguous along key
#pragma unroll
    for (int t = 0; t < 8; ++t) {
      union { bf16x16 v; unsigned u[8]; } vf;
      int d = t * 16 + lr;
#pragma unroll
      for (int p = 0; p < 8; ++p) {
        int key = 2 * p + 16 * half;
        vf.u[p] = Vu[(d * 32 + key) >> 1];
      }
      o[t] = __builtin_amdgcn_wmma_f32_16x16x32_bf16(
          false, pf.v, false, vf.v, (short)0, o[t], false, false);
    }
  }

  // epilogue: O /= l, store merged-heads bf16 attn[s, h*D + d]
#pragma unroll
  for (int t = 0; t < 8; ++t)
#pragma unroll
    for (int r = 0; r < 8; ++r) {
      int m = qw + r + 8 * half;
      float v = o[t][r] / lrow[r];
      Ob[(size_t)m * kE + h * kD + t * 16 + lr] = f32_to_bf16(v);
    }
}

// ---------------------------------------------------------------------------
extern "C" void kernel_launch(void* const* d_in, const int* in_sizes, int n_in,
                              void* d_out, int out_size, void* d_ws, size_t ws_size,
                              hipStream_t stream) {
  (void)in_sizes; (void)n_in; (void)out_size; (void)ws_size;
  const float* hs = (const float*)d_in[0];
  const float* wq = (const float*)d_in[1];
  const float* wk = (const float*)d_in[2];
  const float* wv = (const float*)d_in[3];
  const float* wo = (const float*)d_in[4];
  float* out = (float*)d_out;

  const size_t NE = (size_t)kS * kE;
  unsigned short* p   = (unsigned short*)d_ws;   // 9 bf16 matrices = 72 MB
  unsigned short* hid = p + 0 * NE;
  unsigned short* wqT = p + 1 * NE;              // weights stored [n][k]
  unsigned short* wkT = p + 2 * NE;
  unsigned short* wvT = p + 3 * NE;
  unsigned short* woT = p + 4 * NE;
  unsigned short* Qb  = p + 5 * NE;              // [s][e]
  unsigned short* Kb  = p + 6 * NE;              // [s][e]
  unsigned short* VbT = p + 7 * NE;              // [e][s]  (transposed)
  unsigned short* Ab  = p + 8 * NE;              // [s][e]

  cvt_f32_bf16<<<(int)(NE / 1024), 256, 0, stream>>>(hs, hid);
  dim3 tg(kE / 32, kE / 32);
  cvt_transpose_f32_bf16<<<tg, 256, 0, stream>>>(wq, wqT);
  cvt_transpose_f32_bf16<<<tg, 256, 0, stream>>>(wk, wkT);
  cvt_transpose_f32_bf16<<<tg, 256, 0, stream>>>(wv, wvT);
  cvt_transpose_f32_bf16<<<tg, 256, 0, stream>>>(wo, woT);

  dim3 gg(kS / 128, kE / 128);
  gemm_bf16_wmma<0><<<gg, 256, 0, stream>>>(hid, wqT, Qb);
  gemm_bf16_wmma<0><<<gg, 256, 0, stream>>>(hid, wkT, Kb);
  gemm_bf16_wmma<2><<<gg, 256, 0, stream>>>(hid, wvT, VbT);

  flash_attn_bf16<<<dim3(kS / 64, kH), 128, 0, stream>>>(Qb, Kb, VbT, Ab);

  gemm_bf16_wmma<1><<<gg, 256, 0, stream>>>(Ab, woT, out);
}
